// MultiHeadAttention_59725815218881
// MI455X (gfx1250) — matrix operand order repaired and checked by reading, not compiled, not run on hardware
//
#include <hip/hip_runtime.h>

typedef __attribute__((ext_vector_type(16))) _Float16 v16h;
typedef __attribute__((ext_vector_type(8)))  _Float16 v8h;
typedef __attribute__((ext_vector_type(8)))  float    v8f;

constexpr int Bsz = 4;
constexpr int T   = 2048;
constexpr int D   = 1024;
constexpr int H   = 16;
constexpr int HD  = 64;
constexpr int BT  = Bsz * T;      // 8192
constexpr int N3  = 3 * D;        // 3072

__device__ __forceinline__ v8f wmma_f16(v16h a, v16h b, v8f c) {
  return __builtin_amdgcn_wmma_f32_16x16x32_f16(
      /*neg_a=*/false, a, /*neg_b=*/false, b,
      /*c_mod=*/(short)0, c, /*reuse_a=*/false, /*reuse_b=*/false);
}

// A-matrix 16x32 f16 tile. base -> element [0][0], rows contiguous-K, stride ld.
// Lane l<16: row l, K {0..7,16..23}; lane l+16: row l, K {8..15,24..31}.
__device__ __forceinline__ v16h ld_a(const _Float16* base, int ld) {
  const int lane = threadIdx.x & 31;
  const int row  = lane & 15;
  const int k0   = (lane & 16) ? 8 : 0;
  const _Float16* p = base + (size_t)row * ld + k0;
  v8h lo = *(const v8h*)(p);
  v8h hi = *(const v8h*)(p + 16);
  v16h r;
#pragma unroll
  for (int i = 0; i < 8; ++i) { r[i] = lo[i]; r[i + 8] = hi[i]; }
  return r;
}

// B-matrix 32x16 f16 tile from TRANSPOSED storage: baseT -> [n][k], stride ld.
// Lane n<16: col n, K {0..7,16..23}; lane n+16: col n, K {8..15,24..31}.
__device__ __forceinline__ v16h ld_b(const _Float16* baseT, int ld) {
  const int lane = threadIdx.x & 31;
  const int n    = lane & 15;
  const int k0   = (lane & 16) ? 8 : 0;
  const _Float16* p = baseT + (size_t)n * ld + k0;
  v8h lo = *(const v8h*)(p);
  v8h hi = *(const v8h*)(p + 16);
  v16h r;
#pragma unroll
  for (int i = 0; i < 8; ++i) { r[i] = lo[i]; r[i + 8] = hi[i]; }
  return r;
}

__device__ __forceinline__ float rowmax16(float v) {
#pragma unroll
  for (int m = 1; m < 16; m <<= 1) v = fmaxf(v, __shfl_xor(v, m, 32));
  return v;
}
__device__ __forceinline__ float rowsum16(float v) {
#pragma unroll
  for (int m = 1; m < 16; m <<= 1) v += __shfl_xor(v, m, 32);
  return v;
}

// ---------------- conversion kernels ----------------
__global__ __launch_bounds__(256) void cvt_f16(const float* __restrict__ in,
                                               _Float16* __restrict__ out, int n) {
  int i = blockIdx.x * 256 + threadIdx.x;
  if (i < n) out[i] = (_Float16)in[i];
}

// in: [Kd][Nd] f32  ->  outT: [Nd][Kd] f16
__global__ __launch_bounds__(256) void cvt_f16_T(const float* __restrict__ in,
                                                 _Float16* __restrict__ outT,
                                                 int Kd, int Nd) {
  int i = blockIdx.x * 256 + threadIdx.x;
  if (i < Kd * Nd) {
    int n = i / Kd;
    int k = i - n * Kd;
    outT[i] = (_Float16)in[(size_t)k * Nd + n];
  }
}

// ---------------- QKV GEMM: 64x64 tile per wave ----------------
// [BT,D] x [D,3D] + b -> Q,K in (b,h,t,hd) and V^T in (b,h,hd,t)
__global__ __launch_bounds__(32) void qkv_gemm(const _Float16* __restrict__ xh,
                                               const _Float16* __restrict__ wT,
                                               const float* __restrict__ bias,
                                               _Float16* __restrict__ Qb,
                                               _Float16* __restrict__ Kb,
                                               _Float16* __restrict__ Vt) {
  const int lane = threadIdx.x;
  const int m0 = blockIdx.y * 64;
  const int n0 = blockIdx.x * 64;
  v8f acc[4][4] = {};
  for (int k0 = 0; k0 < D; k0 += 32) {
    v16h a[4];
#pragma unroll
    for (int mt = 0; mt < 4; ++mt)
      a[mt] = ld_a(xh + (size_t)(m0 + mt * 16) * D + k0, D);
#pragma unroll
    for (int j = 0; j < 4; ++j) {
      v16h bt = ld_b(wT + (size_t)(n0 + j * 16) * D + k0, D);
#pragma unroll
      for (int mt = 0; mt < 4; ++mt)
        acc[mt][j] = wmma_f16(a[mt], bt, acc[mt][j]);
    }
  }
  const int col = lane & 15;
  const int mof = (lane & 16) ? 8 : 0;
#pragma unroll
  for (int mt = 0; mt < 4; ++mt) {
#pragma unroll
    for (int j = 0; j < 4; ++j) {
      const int n = n0 + j * 16 + col;
      const float bv = bias[n];
      const int which = n >> 10;        // 0=Q 1=K 2=V
      const int nn = n & (D - 1);
      const int h = nn >> 6, hd = nn & 63;
#pragma unroll
      for (int r = 0; r < 8; ++r) {
        const int m = m0 + mt * 16 + r + mof;
        const int b = m >> 11;          // /T
        const int t = m & (T - 1);
        const size_t bh = (size_t)b * H + h;
        const _Float16 v = (_Float16)(acc[mt][j][r] + bv);
        if (which == 0)      Qb[(bh * T + t) * HD + hd] = v;
        else if (which == 1) Kb[(bh * T + t) * HD + hd] = v;
        else                 Vt[(bh * HD + hd) * T + t] = v;
      }
    }
  }
}

// ---------------- causal flash attention: one wave per 32-query tile ----------------
__global__ __launch_bounds__(32) void flash_attn(const _Float16* __restrict__ Qb,
                                                 const _Float16* __restrict__ Kb,
                                                 const _Float16* __restrict__ Vt,
                                                 _Float16* __restrict__ O) {
  __shared__ __align__(16) _Float16 plds[16 * 32];
  const int lane = threadIdx.x;
  const int q0 = blockIdx.x * 32;
  const int bh = blockIdx.y;
  const int b = bh >> 4, h = bh & 15;
  const int col = lane & 15;
  const int mof = (lane & 16) ? 8 : 0;

  const _Float16* qbase = Qb + ((size_t)bh * T + q0) * HD;
  v16h aq[2][2];
#pragma unroll
  for (int qs = 0; qs < 2; ++qs) {
    aq[qs][0] = ld_a(qbase + (size_t)qs * 16 * HD, HD);       // hd 0..31
    aq[qs][1] = ld_a(qbase + (size_t)qs * 16 * HD + 32, HD);  // hd 32..63
  }

  float m_i[2][8], l_i[2][8];
#pragma unroll
  for (int qs = 0; qs < 2; ++qs)
#pragma unroll
    for (int r = 0; r < 8; ++r) { m_i[qs][r] = -1e30f; l_i[qs][r] = 0.f; }
  v8f acc[2][4] = {};

  const _Float16* vbase = Vt + (size_t)bh * HD * T;
  const int nkb = (q0 >> 5) + 1;              // 32-key blocks up to diagonal
  for (int kb = 0; kb < nkb; ++kb) {
    const int k0 = kb * 32;
    const _Float16* kptr = Kb + ((size_t)bh * T + k0) * HD;
    v16h bk[2][2];                            // K tiles shared by both q-subtiles
#pragma unroll
    for (int t = 0; t < 2; ++t) {
      bk[t][0] = ld_b(kptr + (size_t)t * 16 * HD, HD);
      bk[t][1] = ld_b(kptr + (size_t)t * 16 * HD + 32, HD);
    }
#pragma unroll
    for (int qs = 0; qs < 2; ++qs) {
      const int qr0 = q0 + qs * 16;
      v8f st[2];
#pragma unroll
      for (int t = 0; t < 2; ++t) {
        v8f s = {};
        s = wmma_f16(aq[qs][0], bk[t][0], s);
        s = wmma_f16(aq[qs][1], bk[t][1], s);
        const int c0 = k0 + t * 16;
#pragma unroll
        for (int r = 0; r < 8; ++r) {
          float sv = s[r] * 0.125f;                         // 1/sqrt(64)
          if (c0 >= qr0 && (c0 + col) > (qr0 + r + mof)) sv = -1e30f;
          s[r] = sv;
        }
        st[t] = s;
      }
      // online softmax over the 32 new columns (rows live in 16-lane halves)
#pragma unroll
      for (int r = 0; r < 8; ++r) {
        const float s0 = st[0][r], s1 = st[1][r];
        const float mx = rowmax16(fmaxf(s0, s1));
        const float mn = fmaxf(m_i[qs][r], mx);
        const float p0 = __expf(s0 - mn);
        const float p1 = __expf(s1 - mn);
        const float corr = __expf(m_i[qs][r] - mn);
        const float rs = rowsum16(p0 + p1);
        l_i[qs][r] = l_i[qs][r] * corr + rs;
        m_i[qs][r] = mn;
#pragma unroll
        for (int nt = 0; nt < 4; ++nt) acc[qs][nt][r] *= corr;
        st[0][r] = p0; st[1][r] = p1;
      }
      // C-layout P -> A-layout f16 via LDS
#pragma unroll
      for (int r = 0; r < 8; ++r) {
        plds[(r + mof) * 32 + col]      = (_Float16)st[0][r];
        plds[(r + mof) * 32 + 16 + col] = (_Float16)st[1][r];
      }
      __syncthreads();
      v16h ap = ld_a(plds, 32);
      __syncthreads();
#pragma unroll
      for (int nt = 0; nt < 4; ++nt) {
        v16h bv = ld_b(vbase + (size_t)(nt * 16) * T + k0, T);
        acc[qs][nt] = wmma_f16(ap, bv, acc[qs][nt]);
      }
    }
  }
  // epilogue: O[b,t, h*64+hd] = acc / l
#pragma unroll
  for (int qs = 0; qs < 2; ++qs) {
#pragma unroll
    for (int r = 0; r < 8; ++r) {
      const float invl = 1.0f / l_i[qs][r];
      const size_t orow =
          ((size_t)b * T + q0 + qs * 16 + r + mof) * D + (size_t)h * HD;
#pragma unroll
      for (int nt = 0; nt < 4; ++nt)
        O[orow + nt * 16 + col] = (_Float16)(acc[qs][nt][r] * invl);
    }
  }
}

// ---------------- projection GEMM: 64x64 tile per wave, f32 out + bias ----------------
__global__ __launch_bounds__(32) void proj_gemm(const _Float16* __restrict__ A,
                                                const _Float16* __restrict__ wT,
                                                const float* __restrict__ bias,
                                                float* __restrict__ out) {
  const int lane = threadIdx.x;
  const int m0 = blockIdx.y * 64;
  const int n0 = blockIdx.x * 64;
  v8f acc[4][4] = {};
  for (int k0 = 0; k0 < D; k0 += 32) {
    v16h a[4];
#pragma unroll
    for (int mt = 0; mt < 4; ++mt)
      a[mt] = ld_a(A + (size_t)(m0 + mt * 16) * D + k0, D);
#pragma unroll
    for (int j = 0; j < 4; ++j) {
      v16h bt = ld_b(wT + (size_t)(n0 + j * 16) * D + k0, D);
#pragma unroll
      for (int mt = 0; mt < 4; ++mt)
        acc[mt][j] = wmma_f16(a[mt], bt, acc[mt][j]);
    }
  }
  const int col = lane & 15;
  const int mof = (lane & 16) ? 8 : 0;
#pragma unroll
  for (int mt = 0; mt < 4; ++mt) {
#pragma unroll
    for (int j = 0; j < 4; ++j) {
      const int n = n0 + j * 16 + col;
      const float bv = bias[n];
#pragma unroll
      for (int r = 0; r < 8; ++r) {
        const int m = m0 + mt * 16 + r + mof;
        out[(size_t)m * D + n] = acc[mt][j][r] + bv;
      }
    }
  }
}

extern "C" void kernel_launch(void* const* d_in, const int* in_sizes, int n_in,
                              void* d_out, int out_size, void* d_ws, size_t ws_size,
                              hipStream_t stream) {
  (void)in_sizes; (void)n_in; (void)out_size; (void)ws_size;
  const float* x      = (const float*)d_in[0];
  const float* w_qkv  = (const float*)d_in[1];
  const float* b_qkv  = (const float*)d_in[2];
  const float* w_proj = (const float*)d_in[3];
  const float* b_proj = (const float*)d_in[4];
  float* out = (float*)d_out;

  char* p = (char*)d_ws;
  auto alloc = [&](size_t bytes) {
    void* r = (void*)p;
    p += (bytes + 255) & ~(size_t)255;
    return r;
  };
  _Float16* xh     = (_Float16*)alloc((size_t)BT * D * 2);
  _Float16* wqkvT  = (_Float16*)alloc((size_t)N3 * D * 2);
  _Float16* wprojT = (_Float16*)alloc((size_t)D * D * 2);
  _Float16* Qb     = (_Float16*)alloc((size_t)BT * D * 2);
  _Float16* Kb     = (_Float16*)alloc((size_t)BT * D * 2);
  _Float16* Vt     = (_Float16*)alloc((size_t)BT * D * 2);
  _Float16* att    = (_Float16*)alloc((size_t)BT * D * 2);

  const int nx = BT * D;          // 8,388,608
  cvt_f16<<<(nx + 255) / 256, 256, 0, stream>>>(x, xh, nx);
  const int nw1 = D * N3;         // 3,145,728
  cvt_f16_T<<<(nw1 + 255) / 256, 256, 0, stream>>>(w_qkv, wqkvT, D, N3);
  const int nw2 = D * D;
  cvt_f16_T<<<(nw2 + 255) / 256, 256, 0, stream>>>(w_proj, wprojT, D, D);

  qkv_gemm<<<dim3(N3 / 64, BT / 64), 32, 0, stream>>>(xh, wqkvT, b_qkv, Qb, Kb, Vt);
  flash_attn<<<dim3(T / 32, Bsz * H), 32, 0, stream>>>(Qb, Kb, Vt, att);
  proj_gemm<<<dim3(D / 64, BT / 64), 32, 0, stream>>>(att, wprojT, b_proj, out);
}
